// paramnet_83408264888937
// MI455X (gfx1250) — compile-verified, matrix operand order checked by reading, not run on hardware
//
#include <hip/hip_runtime.h>
#include <hip/hip_bf16.h>

// ---------------------------------------------------------------------------
// CDNA5 (gfx1250, wave32) implementation of the PointNet++-style forward pass.
// All linear layers run through v_wmma_f32_16x16x32_f16 (f16 in, f32 acc).
// Each wave computes a 16x32 output tile: one A fragment feeds two WMMAs.
// Irregular stages (kNN top-100, FPS, ball query, softmax) are VALU/LDS code.
// ---------------------------------------------------------------------------

typedef __attribute__((ext_vector_type(16))) _Float16 v16h;
typedef __attribute__((ext_vector_type(8)))  _Float16 v8h;
typedef __attribute__((ext_vector_type(8)))  float    v8f;

// ---------------------------------------------------------------------------
// WMMA GEMM:  out = act(A[M,Kpad] @ Wt^T + bias)
//   A   : f16 row-major, row stride Kpad (Kpad % 32 == 0, M % 16 == 0 rows alloc)
//   Wt  : f16 [Npad, Kpad]  (weight transposed, zero padded)
//   outh: f16 [*, ldo]  (ldo == Npad), optional
//   outf: f32 [*, Nlog] compact, optional (guarded to m<M, n<Nlog)
// One wave per 16x32 tile (two 16x16 accumulators sharing the A fragment),
// 4 waves per block.
// ---------------------------------------------------------------------------
__global__ void __launch_bounds__(128)
k_gemm(const _Float16* __restrict__ A, const _Float16* __restrict__ Wt,
       const float* __restrict__ bias, _Float16* __restrict__ outh,
       float* __restrict__ outf, int M, int Nlog, int Kpad, int ldo,
       int tilesN, int relu)
{
    const int lane = threadIdx.x & 31;
    const int wv   = threadIdx.x >> 5;
    const int Mtiles  = (M + 15) >> 4;
    const int tilesNP = (tilesN + 1) >> 1;          // pairs of n-tiles
    int pair = blockIdx.x * 4 + wv;
    if (pair >= Mtiles * tilesNP) return;           // wave-uniform exit
    const int tm  = pair / tilesNP;
    const int tp  = pair - tm * tilesNP;
    const int tn0 = tp * 2;
    const int tn1 = (tn0 + 1 < tilesN) ? tn0 + 1 : tn0;   // clamp when tilesN odd

    const int r  = lane & 15;
    const int kb = (lane >> 4) << 3;                // 0 or 8

    const _Float16* pa  = A  + (long long)(tm * 16 + r) * Kpad + kb;
    const _Float16* pb0 = Wt + (long long)(tn0 * 16 + r) * Kpad + kb;
    const _Float16* pb1 = Wt + (long long)(tn1 * 16 + r) * Kpad + kb;

    v8f c0 = {}, c1 = {};
    for (int k0 = 0; k0 < Kpad; k0 += 32) {
        v8h al  = *(const v8h*)(pa);
        v8h ah  = *(const v8h*)(pa + 16);
        v8h b0l = *(const v8h*)(pb0);
        v8h b0h = *(const v8h*)(pb0 + 16);
        v8h b1l = *(const v8h*)(pb1);
        v8h b1h = *(const v8h*)(pb1 + 16);
        v16h a  = __builtin_shufflevector(al,  ah,  0,1,2,3,4,5,6,7,8,9,10,11,12,13,14,15);
        v16h b0 = __builtin_shufflevector(b0l, b0h, 0,1,2,3,4,5,6,7,8,9,10,11,12,13,14,15);
        v16h b1 = __builtin_shufflevector(b1l, b1h, 0,1,2,3,4,5,6,7,8,9,10,11,12,13,14,15);
        c0 = __builtin_amdgcn_wmma_f32_16x16x32_f16(false, a, false, b0,
                                                    (short)0, c0, false, false);
        c1 = __builtin_amdgcn_wmma_f32_16x16x32_f16(false, a, false, b1,
                                                    (short)0, c1, false, false);
        pa += 32; pb0 += 32; pb1 += 32;
        __builtin_prefetch(pa  + 32, 0, 1);         // global_prefetch_b8
        __builtin_prefetch(pb0 + 32, 0, 1);
        __builtin_prefetch(pb1 + 32, 0, 1);
    }

    const int n0 = tn0 * 16 + r;
    const int n1 = tn1 * 16 + r;
    const float bv0 = bias[n0];
    const float bv1 = bias[n1];
    const int mb = tm * 16 + ((lane >> 4) << 3);
#pragma unroll
    for (int i = 0; i < 8; ++i) {
        float v0 = c0[i] + bv0;
        float v1 = c1[i] + bv1;
        if (relu) { v0 = fmaxf(v0, 0.f); v1 = fmaxf(v1, 0.f); }
        int m = mb + i;
        if (outh) {
            outh[(long long)m * ldo + n0] = (_Float16)v0;
            outh[(long long)m * ldo + n1] = (_Float16)v1;
        }
        if (outf && m < M) {
            if (n0 < Nlog) outf[(long long)m * Nlog + n0] = v0;
            if (n1 < Nlog && n1 != n0) outf[(long long)m * Nlog + n1] = v1;
        }
    }
}

// Pack W[K,N] f32 -> Wt[Npad,Kpad] f16 (transposed, zero pad); bias -> f32 pad.
__global__ void k_wtpack(const float* __restrict__ Wsrc, const float* __restrict__ bsrc,
                         _Float16* __restrict__ Wt, float* __restrict__ bp,
                         int K, int N, int Kpad, int Npad)
{
    int id = blockIdx.x * 256 + threadIdx.x;
    int tot = Npad * Kpad;
    if (id < tot) {
        int n = id / Kpad, k = id - n * Kpad;
        float v = (n < N && k < K) ? Wsrc[k * N + n] : 0.f;
        Wt[id] = (_Float16)v;
    }
    if (id < Npad) bp[id] = (id < N) ? bsrc[id] : 0.f;
}

// kNN: per (b,i) block, distances to 2048 points in LDS, select 100 smallest.
__global__ void __launch_bounds__(256)
k_knn(const float* __restrict__ pts, int* __restrict__ nbr)
{
    __shared__ float d[2048];
    __shared__ float rv[256];
    __shared__ int   ri[256];
    const int gid = blockIdx.x;                 // b*2048 + i
    const int b   = gid >> 11;
    const float* P = pts + ((size_t)b << 11) * 3;
    const float qx = pts[(size_t)gid * 3 + 0];
    const float qy = pts[(size_t)gid * 3 + 1];
    const float qz = pts[(size_t)gid * 3 + 2];
    const int t = threadIdx.x;
    for (int j = t; j < 2048; j += 256) {
        float dx = P[j*3] - qx, dy = P[j*3+1] - qy, dz = P[j*3+2] - qz;
        d[j] = dx*dx + dy*dy + dz*dz;
    }
    __syncthreads();
    for (int k = 0; k < 100; ++k) {
        float bv = 3.4e38f; int bi = 0x7fffffff;
        for (int j = t; j < 2048; j += 256) {
            float v = d[j];
            if (v < bv) { bv = v; bi = j; }
        }
        rv[t] = bv; ri[t] = bi;
        __syncthreads();
        for (int o = 128; o > 0; o >>= 1) {
            if (t < o) {
                if (rv[t+o] < rv[t] || (rv[t+o] == rv[t] && ri[t+o] < ri[t])) {
                    rv[t] = rv[t+o]; ri[t] = ri[t+o];
                }
            }
            __syncthreads();
        }
        int sel = ri[0];
        if (t == 0) { nbr[(size_t)gid * 100 + k] = sel; d[sel] = 3.4e38f; }
        __syncthreads();
    }
}

// neigh[g,k,:] = pts[b, nbr[baseG+g, k], :]
__global__ void k_gather_neigh(const float* __restrict__ pts, const int* __restrict__ nbr,
                               float* __restrict__ neigh, int baseG, int G, int K, int NP)
{
    long long id = (long long)blockIdx.x * 256 + threadIdx.x;
    long long tot = (long long)G * K * 3;
    if (id >= tot) return;
    int c = (int)(id % 3); long long tk = id / 3;
    int k = (int)(tk % K); int g = (int)(tk / K);
    int gg = baseG + g;
    int b  = gg / NP;
    int j  = nbr[(long long)gg * K + k];
    neigh[id] = pts[((long long)b * NP + j) * 3 + c];
}

// FPS: one block per group, dist array in dynamic LDS, deterministic start 0.
__global__ void __launch_bounds__(256)
k_fps(const float* __restrict__ xyz, int* __restrict__ idx, int N, int S)
{
    extern __shared__ char smem_raw[];
    float* dist = (float*)smem_raw;     // N floats
    float* rv   = dist + N;             // 256
    int*   ri   = (int*)(rv + 256);     // 256
    const int g = blockIdx.x;
    const float* X = xyz + (long long)g * N * 3;
    const int t = threadIdx.x;
    for (int j = t; j < N; j += 256) dist[j] = 1e10f;
    __syncthreads();
    int far = 0;
    for (int s = 0; s < S; ++s) {
        if (t == 0) idx[(long long)g * S + s] = far;
        float cx = X[far*3], cy = X[far*3+1], cz = X[far*3+2];
        float bv = -1.f; int bi = 0;
        for (int j = t; j < N; j += 256) {
            float dx = X[j*3]-cx, dy = X[j*3+1]-cy, dz = X[j*3+2]-cz;
            float nd = fminf(dist[j], dx*dx+dy*dy+dz*dz);
            dist[j] = nd;
            if (nd > bv) { bv = nd; bi = j; }
        }
        rv[t] = bv; ri[t] = bi;
        __syncthreads();
        for (int o = 128; o > 0; o >>= 1) {
            if (t < o) {
                if (rv[t+o] > rv[t] || (rv[t+o] == rv[t] && ri[t+o] < ri[t])) {
                    rv[t] = rv[t+o]; ri[t] = ri[t+o];
                }
            }
            __syncthreads();
        }
        far = ri[0];
        __syncthreads();
    }
}

// out[g,s,:] = X[g, idx[g,s], :]
__global__ void k_gather3(const float* __restrict__ X, const int* __restrict__ idx,
                          float* __restrict__ out, int G, int N, int S)
{
    long long id = (long long)blockIdx.x * 256 + threadIdx.x;
    long long tot = (long long)G * S * 3;
    if (id >= tot) return;
    int c = (int)(id % 3); long long ts = id / 3;
    int s = (int)(ts % S); int g = (int)(ts / S);
    out[id] = X[((long long)g * N + idx[(long long)g * S + s]) * 3 + c];
}

// Ball query: ascending index scan, first `ns` in-radius; pad with first hit.
__global__ void k_ballq(const float* __restrict__ xyz, const float* __restrict__ nxyz,
                        int* __restrict__ gidx, int G, int N, int S, int ns, float r2)
{
    long long id = (long long)blockIdx.x * 256 + threadIdx.x;
    if (id >= (long long)G * S) return;
    int g = (int)(id / S);
    const float* X = xyz + (long long)g * N * 3;
    const float qx = nxyz[id*3], qy = nxyz[id*3+1], qz = nxyz[id*3+2];
    int* out = gidx + id * ns;
    int cnt = 0, first = -1;
    for (int j = 0; j < N && cnt < ns; ++j) {
        float dx = X[j*3]-qx, dy = X[j*3+1]-qy, dz = X[j*3+2]-qz;
        if (dx*dx + dy*dy + dz*dz <= r2) {
            if (first < 0) first = j;
            out[cnt++] = j;
        }
    }
    if (first < 0) first = 0;
    for (; cnt < ns; ++cnt) out[cnt] = first;
}

// A row (g,s,k): [xyz[j]-new_xyz | feats[j] | 0 pad], f16, stride Kpad.
__global__ void k_buildA(const float* __restrict__ xyz, const float* __restrict__ nxyz,
                         const int* __restrict__ gidx, const _Float16* __restrict__ feats,
                         int fstride, int Cf, _Float16* __restrict__ A,
                         int G, int N, int S, int ns, int Kpad)
{
    long long id = (long long)blockIdx.x * 256 + threadIdx.x;
    long long rows = (long long)G * S * ns;
    if (id >= rows * Kpad) return;
    int col = (int)(id % Kpad); long long row = id / Kpad;
    long long gs = row / ns;
    int g = (int)(gs / S);
    int j = gidx[row];
    _Float16 v = (_Float16)0.f;
    if (col < 3)
        v = (_Float16)(xyz[((long long)g*N + j)*3 + col] - nxyz[gs*3 + col]);
    else if (col < 3 + Cf)
        v = feats[((long long)g*N + j)*fstride + (col - 3)];
    A[id] = v;
}

// group-all A row (g,j): [xyz | feats | 0 pad]
__global__ void k_buildAll(const float* __restrict__ xyz, const _Float16* __restrict__ feats,
                           int fstride, int Cf, _Float16* __restrict__ A,
                           int G, int N, int Kpad)
{
    long long id = (long long)blockIdx.x * 256 + threadIdx.x;
    long long rows = (long long)G * N;
    if (id >= rows * Kpad) return;
    int col = (int)(id % Kpad); long long row = id / Kpad;
    _Float16 v = (_Float16)0.f;
    if (col < 3)            v = (_Float16)xyz[row*3 + col];
    else if (col < 3 + Cf)  v = feats[row*fstride + (col - 3)];
    A[id] = v;
}

// max over ns consecutive rows; in/out stride C.
__global__ void k_maxpool(const _Float16* __restrict__ in, _Float16* __restrict__ out,
                          long long GS, int ns, int C)
{
    long long id = (long long)blockIdx.x * 256 + threadIdx.x;
    if (id >= GS * C) return;
    int c = (int)(id % C); long long gs = id / C;
    const _Float16* p = in + gs * (long long)ns * C + c;
    float m = -3.0e38f;
    for (int j = 0; j < ns; ++j) m = fmaxf(m, (float)p[(long long)j * C]);
    out[gs * C + c] = (_Float16)m;
}

// attr[p, 0:32] = [pts(3) | eula(3) | softmax(edge)(2) | softmax(meta)(13) | 0]
__global__ void k_attr(const float* __restrict__ pts, const float* __restrict__ eula,
                       const float* __restrict__ edge, const float* __restrict__ meta,
                       _Float16* __restrict__ attr, int BN)
{
    int p = blockIdx.x * 256 + threadIdx.x;
    if (p >= BN) return;
    _Float16* o = attr + (long long)p * 32;
    for (int c = 0; c < 3; ++c) o[c]     = (_Float16)pts [(long long)p*3 + c];
    for (int c = 0; c < 3; ++c) o[3 + c] = (_Float16)eula[(long long)p*3 + c];
    float e0 = edge[(long long)p*2], e1 = edge[(long long)p*2 + 1];
    float em = fmaxf(e0, e1);
    float x0 = __expf(e0 - em), x1 = __expf(e1 - em);
    float es = x0 + x1;
    o[6] = (_Float16)(x0 / es); o[7] = (_Float16)(x1 / es);
    float mm = -3.0e38f;
    for (int c = 0; c < 13; ++c) mm = fmaxf(mm, meta[(long long)p*13 + c]);
    float ms = 0.f; float mv[13];
    for (int c = 0; c < 13; ++c) { mv[c] = __expf(meta[(long long)p*13 + c] - mm); ms += mv[c]; }
    for (int c = 0; c < 13; ++c) o[8 + c] = (_Float16)(mv[c] / ms);
    for (int c = 21; c < 32; ++c) o[c] = (_Float16)0.f;
}

__global__ void k_logsm2(const float* __restrict__ in, float* __restrict__ out)
{
    int i = threadIdx.x;
    if (i >= 8) return;
    float a = in[i*2], b = in[i*2+1];
    float m = fmaxf(a, b);
    float l = m + logf(expf(a - m) + expf(b - m));
    out[i*2] = a - l; out[i*2+1] = b - l;
}

// ---------------------------------------------------------------------------
// Host orchestration
// ---------------------------------------------------------------------------
struct Lyr { int wi; int K, N, Kpad, Npad; };
static const Lyr g_lyr[35] = {
    {1,3,16,32,32},{3,16,16,32,32},{5,16,32,32,32},                  // bb0  0..2
    {7,35,32,64,32},{9,32,32,32,32},{11,32,64,32,64},                // bb1  3..5
    {13,67,64,96,64},{15,64,128,64,128},{17,128,256,128,256},        // bb2  6..8
    {19,21,64,32,64},{21,64,64,64,64},{23,64,128,64,128},            // cls0 9..11
    {25,131,128,160,128},{27,128,128,128,128},{29,128,256,128,256},  // cls1 12..14
    {31,259,256,288,256},{33,256,512,256,512},{35,512,1024,512,1024},// cls2 15..17
    {37,256,256,256,256},{39,256,128,256,128},{41,128,32,128,32},{43,32,2,32,16},   // edge 18..21
    {45,256,256,256,256},{47,256,128,256,128},{49,128,32,128,32},{51,32,3,32,16},   // eula 22..25
    {53,1024,512,1024,512},{55,512,256,512,256},{57,256,128,256,128},
    {59,128,64,128,64},{61,64,2,64,16},                              // fc   26..30
    {63,256,256,256,256},{65,256,128,256,128},{67,128,64,128,64},{69,64,13,64,16},  // meta 31..34
};

static inline int nblk(long long tot) { return (int)((tot + 255) / 256); }

extern "C" void kernel_launch(void* const* d_in, const int* in_sizes, int n_in,
                              void* d_out, int out_size, void* d_ws, size_t ws_size,
                              hipStream_t stream)
{
    (void)in_sizes; (void)n_in; (void)out_size; (void)ws_size;
    const float* pts = (const float*)d_in[0];
    float* dout = (float*)d_out;
    char* W = (char*)d_ws;

    // --- workspace bump allocator ---------------------------------------
    size_t off = 0;
    auto A256 = [&](size_t b) { size_t o = off; off = (off + b + 255) & ~(size_t)255; return o; };

    size_t wtOff[35], bOff[35];
    for (int i = 0; i < 35; ++i) {
        wtOff[i] = A256((size_t)g_lyr[i].Npad * g_lyr[i].Kpad * 2);
        bOff[i]  = A256((size_t)g_lyr[i].Npad * 4);
    }
    const size_t nbrO  = A256(16384ULL * 100 * 4);
    const size_t exO   = A256(16384ULL * 256 * 2);
    const size_t hAO   = A256(16384ULL * 256 * 2);
    const size_t hBO   = A256(16384ULL * 256 * 2);
    const size_t attrO = A256(16384ULL * 32 * 2);
    const size_t fxc1O = A256(8ULL * 512 * 4);
    const size_t nxc1O = A256(8ULL * 512 * 3 * 4);
    const size_t gxc1O = A256(8ULL * 512 * 32 * 4);
    const size_t ac1O  = A256(131072ULL * 32 * 2);
    const size_t tcAO  = A256(131072ULL * 128 * 2);
    const size_t tcBO  = A256(131072ULL * 128 * 2);
    const size_t fe1O  = A256(4096ULL * 128 * 2);
    const size_t fxc2O = A256(8ULL * 128 * 4);
    const size_t nxc2O = A256(8ULL * 128 * 3 * 4);
    const size_t gxc2O = A256(8ULL * 128 * 64 * 4);
    const size_t ac2O  = A256(65536ULL * 160 * 2);
    const size_t fe2O  = A256(1024ULL * 256 * 2);
    const size_t ac3O  = A256(1024ULL * 288 * 2);
    const size_t afcO  = A256(16ULL * 1024 * 2);
    const size_t fcoO  = A256(16ULL * 2 * 4);
    // per-chunk arena (CH neighborhoods per chunk)
    const int CH = 512;
    const size_t neighO = A256((size_t)CH * 100 * 3 * 4);
    const size_t fx1O = A256((size_t)CH * 50 * 4);
    const size_t nx1O = A256((size_t)CH * 50 * 3 * 4);
    const size_t gx1O = A256((size_t)CH * 50 * 9 * 4);
    const size_t a1O  = A256((size_t)CH * 50 * 9 * 32 * 2);
    const size_t t1O  = A256((size_t)CH * 50 * 9 * 32 * 2);
    const size_t t2O  = A256((size_t)CH * 50 * 9 * 32 * 2);
    const size_t f1O  = A256((size_t)CH * 50 * 32 * 2);
    const size_t fx2O = A256((size_t)CH * 25 * 4);
    const size_t nx2O = A256((size_t)CH * 25 * 3 * 4);
    const size_t gx2O = A256((size_t)CH * 25 * 12 * 4);
    const size_t a2O  = A256((size_t)CH * 25 * 12 * 64 * 2);
    const size_t u1O  = A256((size_t)CH * 25 * 12 * 64 * 2);
    const size_t u2O  = A256((size_t)CH * 25 * 12 * 64 * 2);
    const size_t f2O  = A256((size_t)CH * 25 * 64 * 2);
    const size_t a3O  = A256((size_t)CH * 25 * 96 * 2);
    const size_t v1O  = A256((size_t)CH * 25 * 256 * 2);
    const size_t v2O  = A256((size_t)CH * 25 * 256 * 2);

    auto GEMM = [&](const void* A, int li, void* outh, float* outf, int M, int ldo, int relu) {
        const Lyr& L = g_lyr[li];
        int tilesN  = L.Npad / 16;
        int tilesNP = (tilesN + 1) / 2;
        int tiles   = ((M + 15) / 16) * tilesNP;
        k_gemm<<<(tiles + 3) / 4, 128, 0, stream>>>(
            (const _Float16*)A, (const _Float16*)(W + wtOff[li]), (const float*)(W + bOff[li]),
            (_Float16*)outh, outf, M, L.N, L.Kpad, ldo, tilesN, relu);
    };

    // --- 0) pack weights to f16 transposed -------------------------------
    for (int i = 0; i < 35; ++i) {
        const Lyr& L = g_lyr[i];
        k_wtpack<<<nblk((long long)L.Npad * L.Kpad), 256, 0, stream>>>(
            (const float*)d_in[L.wi], (const float*)d_in[L.wi + 1],
            (_Float16*)(W + wtOff[i]), (float*)(W + bOff[i]), L.K, L.N, L.Kpad, L.Npad);
    }

    // --- 1) surface kNN (top-100) ----------------------------------------
    k_knn<<<16384, 256, 0, stream>>>(pts, (int*)(W + nbrO));

    // --- 2) per-neighborhood backbone, chunked ---------------------------
    _Float16* ex = (_Float16*)(W + exO);
    for (int c = 0; c < 16384 / CH; ++c) {
        int baseG = c * CH;
        float* neigh = (float*)(W + neighO);
        k_gather_neigh<<<nblk((long long)CH * 100 * 3), 256, 0, stream>>>(
            pts, (const int*)(W + nbrO), neigh, baseG, CH, 100, 2048);
        // SA1: N=100 -> S=50, ns=9, r=0.2, MLP 3->16->16->32
        k_fps<<<CH, 256, (100 + 512) * 4, stream>>>(neigh, (int*)(W + fx1O), 100, 50);
        k_gather3<<<nblk((long long)CH * 50 * 3), 256, 0, stream>>>(
            neigh, (const int*)(W + fx1O), (float*)(W + nx1O), CH, 100, 50);
        k_ballq<<<nblk((long long)CH * 50), 256, 0, stream>>>(
            neigh, (const float*)(W + nx1O), (int*)(W + gx1O), CH, 100, 50, 9, 0.04f);
        k_buildA<<<nblk((long long)CH * 50 * 9 * 32), 256, 0, stream>>>(
            neigh, (const float*)(W + nx1O), (const int*)(W + gx1O),
            (const _Float16*)nullptr, 0, 0, (_Float16*)(W + a1O), CH, 100, 50, 9, 32);
        int M1 = CH * 50 * 9;
        GEMM(W + a1O, 0, W + t1O, nullptr, M1, 32, 1);
        GEMM(W + t1O, 1, W + t2O, nullptr, M1, 32, 1);
        GEMM(W + t2O, 2, W + t1O, nullptr, M1, 32, 1);
        k_maxpool<<<nblk((long long)CH * 50 * 32), 256, 0, stream>>>(
            (const _Float16*)(W + t1O), (_Float16*)(W + f1O), (long long)CH * 50, 9, 32);
        // SA2: N=50 -> S=25, ns=12, r=0.4, MLP 35->32->32->64
        k_fps<<<CH, 256, (50 + 512) * 4, stream>>>((const float*)(W + nx1O), (int*)(W + fx2O), 50, 25);
        k_gather3<<<nblk((long long)CH * 25 * 3), 256, 0, stream>>>(
            (const float*)(W + nx1O), (const int*)(W + fx2O), (float*)(W + nx2O), CH, 50, 25);
        k_ballq<<<nblk((long long)CH * 25), 256, 0, stream>>>(
            (const float*)(W + nx1O), (const float*)(W + nx2O), (int*)(W + gx2O), CH, 50, 25, 12, 0.16f);
        k_buildA<<<nblk((long long)CH * 25 * 12 * 64), 256, 0, stream>>>(
            (const float*)(W + nx1O), (const float*)(W + nx2O), (const int*)(W + gx2O),
            (const _Float16*)(W + f1O), 32, 32, (_Float16*)(W + a2O), CH, 50, 25, 12, 64);
        int M2 = CH * 25 * 12;
        GEMM(W + a2O, 3, W + u1O, nullptr, M2, 32, 1);
        GEMM(W + u1O, 4, W + u2O, nullptr, M2, 32, 1);
        GEMM(W + u2O, 5, W + u1O, nullptr, M2, 64, 1);
        k_maxpool<<<nblk((long long)CH * 25 * 64), 256, 0, stream>>>(
            (const _Float16*)(W + u1O), (_Float16*)(W + f2O), (long long)CH * 25, 12, 64);
        // SA3 group-all: N=25, MLP 67->64->128->256 -> ex
        k_buildAll<<<nblk((long long)CH * 25 * 96), 256, 0, stream>>>(
            (const float*)(W + nx2O), (const _Float16*)(W + f2O), 64, 64,
            (_Float16*)(W + a3O), CH, 25, 96);
        int M3 = CH * 25;
        GEMM(W + a3O, 6, W + v1O, nullptr, M3, 64, 1);
        GEMM(W + v1O, 7, W + v2O, nullptr, M3, 128, 1);
        GEMM(W + v2O, 8, W + v1O, nullptr, M3, 256, 1);
        k_maxpool<<<nblk((long long)CH * 256), 256, 0, stream>>>(
            (const _Float16*)(W + v1O), ex + (long long)baseG * 256, (long long)CH, 25, 256);
    }

    // --- 3) heads on ex [16384,256]: eula / edge / meta -------------------
    const int BN = 16384;
    auto head = [&](int l0, float* outf) {
        GEMM(ex, l0 + 0, W + hAO, nullptr, BN, 256, 1);
        GEMM(W + hAO, l0 + 1, W + hBO, nullptr, BN, 128, 1);
        GEMM(W + hBO, l0 + 2, W + hAO, nullptr, BN, 32, 1);
        GEMM(W + hAO, l0 + 3, nullptr, outf, BN, 16, 0);
    };
    float* eulaOut = dout + 16;
    float* edgeOut = dout + 16 + 49152;
    float* metaOut = dout + 16 + 49152 + 32768;
    head(22, eulaOut);   // eula
    head(18, edgeOut);   // edge
    head(31, metaOut);   // meta

    // --- 4) attr = [xyz | eula | softmax(edge) | softmax(meta)] ----------
    k_attr<<<nblk(BN), 256, 0, stream>>>(pts, eulaOut, edgeOut, metaOut,
                                         (_Float16*)(W + attrO), BN);

    // --- 5) cls SA1: B=8, N=2048 -> S=512, ns=32, MLP 21->64->64->128 ----
    k_fps<<<8, 256, (2048 + 512) * 4, stream>>>(pts, (int*)(W + fxc1O), 2048, 512);
    k_gather3<<<nblk(8LL * 512 * 3), 256, 0, stream>>>(
        pts, (const int*)(W + fxc1O), (float*)(W + nxc1O), 8, 2048, 512);
    k_ballq<<<nblk(8LL * 512), 256, 0, stream>>>(
        pts, (const float*)(W + nxc1O), (int*)(W + gxc1O), 8, 2048, 512, 32, 0.04f);
    k_buildA<<<nblk(8LL * 512 * 32 * 32), 256, 0, stream>>>(
        pts, (const float*)(W + nxc1O), (const int*)(W + gxc1O),
        (const _Float16*)(W + attrO) + 3, 32, 18, (_Float16*)(W + ac1O), 8, 2048, 512, 32, 32);
    int Mc1 = 8 * 512 * 32;
    GEMM(W + ac1O, 9,  W + tcAO, nullptr, Mc1, 64, 1);
    GEMM(W + tcAO, 10, W + tcBO, nullptr, Mc1, 64, 1);
    GEMM(W + tcBO, 11, W + tcAO, nullptr, Mc1, 128, 1);
    k_maxpool<<<nblk(4096LL * 128), 256, 0, stream>>>(
        (const _Float16*)(W + tcAO), (_Float16*)(W + fe1O), 4096LL, 32, 128);

    // --- 6) cls SA2: N=512 -> S=128, ns=64, MLP 131->128->128->256 -------
    k_fps<<<8, 256, (512 + 512) * 4, stream>>>((const float*)(W + nxc1O), (int*)(W + fxc2O), 512, 128);
    k_gather3<<<nblk(8LL * 128 * 3), 256, 0, stream>>>(
        (const float*)(W + nxc1O), (const int*)(W + fxc2O), (float*)(W + nxc2O), 8, 512, 128);
    k_ballq<<<nblk(8LL * 128), 256, 0, stream>>>(
        (const float*)(W + nxc1O), (const float*)(W + nxc2O), (int*)(W + gxc2O), 8, 512, 128, 64, 0.16f);
    k_buildA<<<nblk(8LL * 128 * 64 * 160), 256, 0, stream>>>(
        (const float*)(W + nxc1O), (const float*)(W + nxc2O), (const int*)(W + gxc2O),
        (const _Float16*)(W + fe1O), 128, 128, (_Float16*)(W + ac2O), 8, 512, 128, 64, 160);
    int Mc2 = 8 * 128 * 64;
    GEMM(W + ac2O, 12, W + tcAO, nullptr, Mc2, 128, 1);
    GEMM(W + tcAO, 13, W + tcBO, nullptr, Mc2, 128, 1);
    GEMM(W + tcBO, 14, W + tcAO, nullptr, Mc2, 256, 1);
    k_maxpool<<<nblk(1024LL * 256), 256, 0, stream>>>(
        (const _Float16*)(W + tcAO), (_Float16*)(W + fe2O), 1024LL, 64, 256);

    // --- 7) cls group-all: N=128, MLP 259->256->512->1024 -> glob --------
    k_buildAll<<<nblk(1024LL * 288), 256, 0, stream>>>(
        (const float*)(W + nxc2O), (const _Float16*)(W + fe2O), 256, 256,
        (_Float16*)(W + ac3O), 8, 128, 288);
    int Mc3 = 8 * 128;
    GEMM(W + ac3O, 15, W + tcAO, nullptr, Mc3, 256, 1);
    GEMM(W + tcAO, 16, W + tcBO, nullptr, Mc3, 512, 1);
    GEMM(W + tcBO, 17, W + tcAO, nullptr, Mc3, 1024, 1);
    k_maxpool<<<nblk(8LL * 1024), 256, 0, stream>>>(
        (const _Float16*)(W + tcAO), (_Float16*)(W + afcO), 8LL, 128, 1024);

    // --- 8) fc head: 1024->512->256->128->64->2, log_softmax ------------
    GEMM(W + afcO, 26, W + tcAO, nullptr, 16, 512, 1);
    GEMM(W + tcAO, 27, W + tcBO, nullptr, 16, 256, 1);
    GEMM(W + tcBO, 28, W + tcAO, nullptr, 16, 128, 1);
    GEMM(W + tcAO, 29, W + tcBO, nullptr, 16, 64, 1);
    GEMM(W + tcBO, 30, nullptr, (float*)(W + fcoO), 8, 16, 0);
    k_logsm2<<<1, 32, 0, stream>>>((const float*)(W + fcoO), dout);
}
